// CausalAttention_v1_26826365731266
// MI455X (gfx1250) — matrix operand order repaired and checked by reading, not compiled, not run on hardware
//
#include <hip/hip_runtime.h>
#include <hip/hip_bf16.h>
#include <math.h>

// ---------------------------------------------------------------------------
// Causal attention, S=4096, d=1024, fp32 I/O.
//   reference == causal_softmax(QK^T/sqrt(d)) @ V   (tril+renorm identity)
// GEMMs on v_wmma_f32_16x16x32_bf16 (fp32 accum), 64x32 register block/wave.
// Scores & PV stages share the A operand block-wide through LDS, staged by
// the Tensor Data Mover (tensor_load_to_lds + s_wait_tensorcnt), double-
// buffered, with TDM-inserted LDS padding for conflict-free ds_load_b128.
// ---------------------------------------------------------------------------

typedef __attribute__((ext_vector_type(16))) __bf16       v16bf;
typedef __attribute__((ext_vector_type(8)))  __bf16       v8bf;
typedef __attribute__((ext_vector_type(8)))  float        v8f;
typedef __attribute__((ext_vector_type(8)))  float        v8fl;
typedef __attribute__((ext_vector_type(4)))  unsigned int v4u;
typedef __attribute__((ext_vector_type(8)))  int          v8i;
typedef __attribute__((ext_vector_type(4)))  int          v4i;

#define S_LEN 4096
#define D_DIM 1024
#define MT    4                 // 16-row subtiles per wave
#define NT    2                 // 16-col subtiles per wave
#define LDS_PITCH_DW 20         // 16 data dwords + 4 pad dwords = 80B rows

#if __has_builtin(__builtin_amdgcn_tensor_load_to_lds) && \
    __has_builtin(__builtin_amdgcn_s_wait_tensorcnt)
#define HAS_TDM 1
#else
#define HAS_TDM 0
#endif

// ---- TDM: DMA a 64-row x 16-dword (64x32 bf16) tile into LDS --------------
// D# per CDNA5 ISA. 2D tile, data_size=4B,
// pad_enable: +4 dwords after every 16 dwords -> 80B LDS row pitch.
__device__ __forceinline__ void tdm_stage_64x32(unsigned int lds_byte_off,
                                                const __bf16* __restrict__ gsrc,
                                                unsigned int row_stride_dw) {
#if HAS_TDM
    const unsigned long long ga = (unsigned long long)(uintptr_t)gsrc;
    v4u g0;
    g0[0] = 1u;                                        // count=1 (valid, user)
    g0[1] = lds_byte_off;                              // lds_addr
    g0[2] = (unsigned int)ga;                          // global_addr[31:0]
    g0[3] = (unsigned int)(ga >> 32) | (2u << 30);     // global_addr[56:32]|type=2
    v8i g1;
    g1[0] = (int)((2u << 16)                           // data_size = 4B
                | (1u << 20)                           // pad_enable
                | (3u << 22)                           // pad_interval = 16 dwords
                | (3u << 25));                         // pad_amount   = 4 dwords
    g1[1] = (int)((row_stride_dw & 0xFFFFu) << 16);    // tensor_dim0[15:0]
    g1[2] = (int)((row_stride_dw >> 16) | (4096u << 16)); // dim0[31:16]|dim1[15:0]
    g1[3] = (int)(16u << 16);                          // dim1[31:16]=0 | tile_dim0=16
    g1[4] = 64;                                        // tile_dim1=64, tile_dim2=0
    g1[5] = (int)row_stride_dw;                        // tensor_dim0_stride[31:0]
    g1[6] = 0;                                         // stride[47:32]|dim1_stride lo
    g1[7] = 0;
    const v4i gz4 = {0, 0, 0, 0};                      // groups 2/3: unused (2D)
    const v8i gz8 = {0, 0, 0, 0, 0, 0, 0, 0};
    // 6-arg toolchain variant: (v4u, v8i, v4i, v4i, v8i, i32 cpol)
    __builtin_amdgcn_tensor_load_to_lds(g0, g1, gz4, gz4, gz8, 0);
#endif
}

__device__ __forceinline__ void stage_tile(unsigned int* __restrict__ lds_tile,
                                           const __bf16* __restrict__ gsrc,
                                           unsigned int row_stride_dw,
                                           int wave, int tid) {
#if HAS_TDM
    if (wave == 0)
        tdm_stage_64x32((unsigned int)(uintptr_t)lds_tile, gsrc, row_stride_dw);
    (void)tid;
#else
    const unsigned int* __restrict__ g32 = (const unsigned int*)gsrc;
    for (int d = tid; d < 64 * 16; d += 256) {
        const int row = d >> 4, col = d & 15;
        lds_tile[row * LDS_PITCH_DW + col] = g32[(size_t)row * row_stride_dw + col];
    }
#endif
}

__device__ __forceinline__ void stage_wait(int wave) {
#if HAS_TDM
    if (wave == 0) __builtin_amdgcn_s_wait_tensorcnt(0);
#endif
    __syncthreads();
}

// ---- fragment loaders (contiguous k = 16*hi + e, same for A and B) -----------
__device__ __forceinline__ v16bf a_frag_f32(const float* __restrict__ row,
                                            int k0, int hi) {
    v8fl lo = *(const v8fl*)(row + k0 + hi * 16);
    v8fl up = *(const v8fl*)(row + k0 + hi * 16 + 8);
    v16bf a;
#pragma unroll
    for (int e = 0; e < 8; ++e) { a[e] = (__bf16)lo[e]; a[8 + e] = (__bf16)up[e]; }
    return a;
}
__device__ __forceinline__ v16bf b_frag_f32(const float* __restrict__ row,
                                            int k0, int hi) {
    v8fl b0 = *(const v8fl*)(row + k0 + hi * 16);
    v8fl b1 = *(const v8fl*)(row + k0 + hi * 16 + 8);
    v16bf b;
#pragma unroll
    for (int e = 0; e < 8; ++e) { b[e] = (__bf16)b0[e]; b[8 + e] = (__bf16)b1[e]; }
    return b;
}
// A fragment from the padded LDS tile (row pitch 80B -> conflict-free banks)
__device__ __forceinline__ v16bf a_frag_lds(const unsigned int* __restrict__ tile,
                                            int row_local, int hi) {
    const __bf16* lrow = (const __bf16*)(tile + (size_t)row_local * LDS_PITCH_DW);
    v8bf lo = *(const v8bf*)(lrow + hi * 16);
    v8bf up = *(const v8bf*)(lrow + hi * 16 + 8);
    v16bf a;
#pragma unroll
    for (int e = 0; e < 8; ++e) { a[e] = lo[e]; a[8 + e] = up[e]; }
    return a;
}
__device__ __forceinline__ v16bf b_frag_bf(const __bf16* __restrict__ row,
                                           int k0, int hi) {
    return *(const v16bf*)(row + k0 + hi * 16);   // 32B contiguous, aligned
}

// ---- Stage 1: projections (x,W f32; convert in-register) ------------------
__global__ void __launch_bounds__(256)
proj_gemm_kernel(const float* __restrict__ x,
                 const float* __restrict__ Wq,
                 const float* __restrict__ Wk,
                 const float* __restrict__ Wv,
                 __bf16* __restrict__ Qb,
                 __bf16* __restrict__ Kb,
                 __bf16* __restrict__ Vt) {
    const int which = blockIdx.y;                  // 0=Q, 1=K, 2=V(transposed)
    const float* __restrict__ W = (which == 0) ? Wq : ((which == 1) ? Wk : Wv);

    const int wave = threadIdx.x >> 5, lane = threadIdx.x & 31;
    const int tile = blockIdx.x * 8 + wave;
    const int tilesN = D_DIM / (16 * NT);          // 32
    const int tm = tile / tilesN, tn = tile % tilesN;
    const int m0 = tm * 16 * MT, n0 = tn * 16 * NT;
    const int r = lane & 15, hi = lane >> 4;

    const float* __restrict__ a0 = x + (size_t)(m0 + r) * D_DIM;
    const float* __restrict__ b0 = W + (size_t)(n0 + r) * D_DIM;

    v8f acc[MT][NT] = {};
    for (int k0 = 0; k0 < D_DIM; k0 += 32) {
        __builtin_prefetch(a0 + k0 + 64, 0, 1);    // global_prefetch_b8
        __builtin_prefetch(b0 + k0 + 64, 0, 1);
        v16bf af[MT], bf[NT];
#pragma unroll
        for (int i = 0; i < MT; ++i)
            af[i] = a_frag_f32(a0 + (size_t)(16 * i) * D_DIM, k0, hi);
#pragma unroll
        for (int j = 0; j < NT; ++j)
            bf[j] = b_frag_f32(b0 + (size_t)(16 * j) * D_DIM, k0, hi);
#pragma unroll
        for (int i = 0; i < MT; ++i)
#pragma unroll
            for (int j = 0; j < NT; ++j)
                acc[i][j] = __builtin_amdgcn_wmma_f32_16x16x32_bf16(
                    false, af[i], false, bf[j], (short)0, acc[i][j], false, false);
    }

    if (which == 2) {
#pragma unroll
        for (int i = 0; i < MT; ++i)
#pragma unroll
            for (int j = 0; j < NT; ++j)
#pragma unroll
                for (int q = 0; q < 8; ++q) {
                    const int m = m0 + 16 * i + q + 8 * hi;
                    const int n = n0 + 16 * j + r;
                    Vt[(size_t)n * S_LEN + m] = (__bf16)acc[i][j][q];
                }
    } else {
        __bf16* __restrict__ out = (which == 0) ? Qb : Kb;
#pragma unroll
        for (int i = 0; i < MT; ++i)
#pragma unroll
            for (int j = 0; j < NT; ++j)
#pragma unroll
                for (int q = 0; q < 8; ++q) {
                    const int m = m0 + 16 * i + q + 8 * hi;
                    const int n = n0 + 16 * j + r;
                    out[(size_t)m * D_DIM + n] = (__bf16)acc[i][j][q];
                }
    }
}

// ---- Stage 2a: scores = (1/32)*Q K^T; A(Q) staged via TDM, double-buffered -
// Block: 64 rows x 256 cols (8 waves x 64x32). Whole-block causal skip.
__global__ void __launch_bounds__(256)
scores_gemm_kernel(const __bf16* __restrict__ Qb,
                   const __bf16* __restrict__ Kb,
                   float* __restrict__ Sc) {
    __shared__ unsigned int atile[2][64 * LDS_PITCH_DW];   // 2 x 5 KB
    const int m0 = blockIdx.y * 64;
    const int nb = blockIdx.x * 256;
    if (nb > m0 + 63) return;                      // block-uniform causal skip
    const int tid = threadIdx.x;
    const int wave = tid >> 5, lane = tid & 31;
    const int n0 = nb + wave * 32;
    const int r = lane & 15, hi = lane >> 4;

    const __bf16* __restrict__ asrc = Qb + (size_t)m0 * D_DIM;
    const __bf16* __restrict__ b0   = Kb + (size_t)(n0 + r) * D_DIM;

    stage_tile(atile[0], asrc, D_DIM / 2, wave, tid);
    stage_wait(wave);

    v8f acc[MT][NT] = {};
    for (int k0 = 0; k0 < D_DIM; k0 += 32) {
        const int cur = (k0 >> 5) & 1;
        if (k0 + 32 < D_DIM)
            stage_tile(atile[cur ^ 1], asrc + (k0 + 32), D_DIM / 2, wave, tid);
        v16bf af[MT], bf[NT];
#pragma unroll
        for (int i = 0; i < MT; ++i)
            af[i] = a_frag_lds(atile[cur], 16 * i + r, hi);
#pragma unroll
        for (int j = 0; j < NT; ++j)
            bf[j] = b_frag_bf(b0 + (size_t)(16 * j) * D_DIM, k0, hi);
#pragma unroll
        for (int i = 0; i < MT; ++i)
#pragma unroll
            for (int j = 0; j < NT; ++j)
                acc[i][j] = __builtin_amdgcn_wmma_f32_16x16x32_bf16(
                    false, af[i], false, bf[j], (short)0, acc[i][j], false, false);
        stage_wait(wave);
    }
    const float scale = 0.03125f;                  // 1/sqrt(1024)
#pragma unroll
    for (int i = 0; i < MT; ++i)
#pragma unroll
        for (int j = 0; j < NT; ++j)
#pragma unroll
            for (int q = 0; q < 8; ++q) {
                const int m = m0 + 16 * i + q + 8 * hi;
                const int n = n0 + 16 * j + r;
                Sc[(size_t)m * S_LEN + n] = acc[i][j][q] * scale;
            }
}

// ---- Stage 2b: causal row softmax -> P (bf16), upper triangle zeroed ------
__global__ void __launch_bounds__(256)
softmax_kernel(const float* __restrict__ Sc, __bf16* __restrict__ P) {
    const int row = blockIdx.x;
    const int tid = threadIdx.x;
    const int len = row + 1;
    const float* __restrict__ src = Sc + (size_t)row * S_LEN;
    __shared__ float red[256];

    float m = -3.4e38f;
    for (int c = tid; c < len; c += 256) m = fmaxf(m, src[c]);
    red[tid] = m; __syncthreads();
#pragma unroll
    for (int s = 128; s > 0; s >>= 1) {
        if (tid < s) red[tid] = fmaxf(red[tid], red[tid + s]);
        __syncthreads();
    }
    m = red[0]; __syncthreads();

    float sum = 0.f;
    for (int c = tid; c < len; c += 256) sum += __expf(src[c] - m);
    red[tid] = sum; __syncthreads();
#pragma unroll
    for (int s = 128; s > 0; s >>= 1) {
        if (tid < s) red[tid] += red[tid + s];
        __syncthreads();
    }
    const float inv = 1.0f / red[0];

    __bf16* __restrict__ dst = P + (size_t)row * S_LEN;
    for (int c = tid; c < S_LEN; c += 256)
        dst[c] = (c < len) ? (__bf16)(__expf(src[c] - m) * inv) : (__bf16)0.0f;
}

// ---- Stage 3: out = P V; A(P) staged via TDM; causal k bound --------------
__global__ void __launch_bounds__(256)
pv_gemm_kernel(const __bf16* __restrict__ P,
               const __bf16* __restrict__ Vt,
               float* __restrict__ out) {
    __shared__ unsigned int atile[2][64 * LDS_PITCH_DW];
    const int m0 = blockIdx.y * 64;
    const int tid = threadIdx.x;
    const int wave = tid >> 5, lane = tid & 31;
    const int n0 = blockIdx.x * 256 + wave * 32;
    const int r = lane & 15, hi = lane >> 4;

    const __bf16* __restrict__ asrc = P  + (size_t)m0 * S_LEN;
    const __bf16* __restrict__ b0   = Vt + (size_t)(n0 + r) * S_LEN;

    stage_tile(atile[0], asrc, S_LEN / 2, wave, tid);
    stage_wait(wave);

    v8f acc[MT][NT] = {};
    const int kend = m0 + 64;                      // P zero beyond diagonal
    for (int k0 = 0; k0 < kend; k0 += 32) {
        const int cur = (k0 >> 5) & 1;
        if (k0 + 32 < kend)
            stage_tile(atile[cur ^ 1], asrc + (k0 + 32), S_LEN / 2, wave, tid);
        v16bf af[MT], bf[NT];
#pragma unroll
        for (int i = 0; i < MT; ++i)
            af[i] = a_frag_lds(atile[cur], 16 * i + r, hi);
#pragma unroll
        for (int j = 0; j < NT; ++j)
            bf[j] = b_frag_bf(b0 + (size_t)(16 * j) * S_LEN, k0, hi);
#pragma unroll
        for (int i = 0; i < MT; ++i)
#pragma unroll
            for (int j = 0; j < NT; ++j)
                acc[i][j] = __builtin_amdgcn_wmma_f32_16x16x32_bf16(
                    false, af[i], false, bf[j], (short)0, acc[i][j], false, false);
        stage_wait(wave);
    }
#pragma unroll
    for (int i = 0; i < MT; ++i)
#pragma unroll
        for (int j = 0; j < NT; ++j)
#pragma unroll
            for (int q = 0; q < 8; ++q) {
                const int m = m0 + 16 * i + q + 8 * hi;
                const int n = n0 + 16 * j + r;
                out[(size_t)m * D_DIM + n] = acc[i][j][q];
            }
}

// ---------------------------------------------------------------------------
extern "C" void kernel_launch(void* const* d_in, const int* in_sizes, int n_in,
                              void* d_out, int out_size, void* d_ws, size_t ws_size,
                              hipStream_t stream) {
    const float* x  = (const float*)d_in[0];
    const float* Wq = (const float*)d_in[1];
    const float* Wk = (const float*)d_in[2];
    const float* Wv = (const float*)d_in[3];
    float* out = (float*)d_out;

    char* ws = (char*)d_ws;
    __bf16* Qb = (__bf16*)(ws);                          //  8 MB
    __bf16* Kb = (__bf16*)(ws + ((size_t)8  << 20));     //  8 MB
    __bf16* Vt = (__bf16*)(ws + ((size_t)16 << 20));     //  8 MB
    float*  Sc = (float*) (ws + ((size_t)24 << 20));     // 64 MB
    __bf16* P  = (__bf16*)(ws + ((size_t)88 << 20));     // 32 MB

    dim3 blk(256, 1, 1);
    const int projWT = (S_LEN / (16 * MT)) * (D_DIM / (16 * NT)); // 2048

    proj_gemm_kernel<<<dim3(projWT / 8, 3, 1), blk, 0, stream>>>(
        x, Wq, Wk, Wv, Qb, Kb, Vt);

    // 64-row x 256-col blocks over the 4096x4096 score matrix
    scores_gemm_kernel<<<dim3(S_LEN / 256, S_LEN / 64, 1), blk, 0, stream>>>(
        Qb, Kb, Sc);

    softmax_kernel<<<dim3(S_LEN, 1, 1), blk, 0, stream>>>(Sc, P);

    // 64-row x 256-col blocks over the 4096x1024 output
    pv_gemm_kernel<<<dim3(D_DIM / 256, S_LEN / 64, 1), blk, 0, stream>>>(
        P, Vt, out);
}
